// LSTMModel_77266461655604
// MI455X (gfx1250) — compile-verified, matrix operand order checked by reading, not run on hardware
//
#include <hip/hip_runtime.h>
#include <math.h>

// ---------------------------------------------------------------------------
// LSTM (2 layers, B=64, T=512, IN=256, H=1024) + FC(1024->256) for gfx1250.
// bf16 WMMA (v_wmma_f32_16x16x32_bf16) with f32 accumulation.
//  - Projection GEMMs: 32x64 wave tiles (1.5 b128 loads per WMMA).
//  - Recurrent scan: split-K=2 across wave pairs, LDS reduction, halving the
//    dependent WMMA chain per timestep (latency-bound part).
// ---------------------------------------------------------------------------

#define BDIM 256              // 8 waves per block (wave32)
static const int B_  = 64;
static const int T_  = 512;
static const int IN_ = 256;
static const int H_  = 1024;
static const int OUT_= 256;
static const int G4_ = 4096;  // 4*H
static const int M_  = 64 * 512; // 32768 rows (time-major [T][B])

typedef __attribute__((ext_vector_type(16))) __bf16          v16bf;
typedef __attribute__((ext_vector_type(8)))  float           v8f;
typedef __attribute__((ext_vector_type(8)))  unsigned short  u16x8;

union FragU { v16bf v; u16x8 h[2]; };

__device__ __forceinline__ unsigned short f2bf(float f) {
  union { float f; unsigned u; } x; x.f = f;
  unsigned u = x.u;
  u += 0x7FFFu + ((u >> 16) & 1u);   // round-to-nearest-even
  return (unsigned short)(u >> 16);
}
__device__ __forceinline__ float bf2f(unsigned short h) {
  union { unsigned u; float f; } x; x.u = ((unsigned)h) << 16;
  return x.f;
}
__device__ __forceinline__ float sigmoidf_(float x) {
  return 1.0f / (1.0f + __expf(-x));
}

// Load a 16x32 bf16 WMMA fragment (A or B operand) from a row-major matrix.
// Lane L: row = row0 + (L&15), K-half select = L>>4.
// elems 0..7  = mat[row][k0 + sel*8 + 0..7]
// elems 8..15 = mat[row][k0 + 16 + sel*8 + 0..7]
// Two contiguous 16-byte (b128) loads per lane.
__device__ __forceinline__ v16bf load_frag(const unsigned short* __restrict__ mat,
                                           int ld, int row0, int k0, int lane) {
  int r   = lane & 15;
  int sel = (lane >> 4) & 1;
  const unsigned short* p = mat + (size_t)(row0 + r) * (size_t)ld + k0 + sel * 8;
  FragU f;
  f.h[0] = *(const u16x8*)(p);
  f.h[1] = *(const u16x8*)(p + 16);
  return f.v;
}

__device__ __forceinline__ v8f wmma_bf16(v16bf a, v16bf b, v8f c) {
  return __builtin_amdgcn_wmma_f32_16x16x32_bf16(false, a, false, b,
                                                 (short)0, c, false, false);
}

// ------------------------------ prep kernels -------------------------------

__global__ void k_cvt(const float* __restrict__ s, unsigned short* __restrict__ d, int n) {
  for (int i = blockIdx.x * blockDim.x + threadIdx.x; i < n; i += gridDim.x * blockDim.x)
    d[i] = f2bf(s[i]);
}

// x: [B=64][T=512][256] fp32  ->  d: [T=512][B=64][256] bf16 (time-major)
__global__ void k_cvt_x_tmajor(const float* __restrict__ x, unsigned short* __restrict__ d) {
  const int n = 64 * 512 * 256;
  for (int i = blockIdx.x * blockDim.x + threadIdx.x; i < n; i += gridDim.x * blockDim.x) {
    int k  = i & 255;
    int bt = i >> 8;
    int t  = bt & 511;
    int b  = bt >> 9;
    d[(((size_t)t << 6) + b) * 256 + k] = f2bf(x[i]);
  }
}

__global__ void k_bias_combine(const float* __restrict__ a, const float* __restrict__ b,
                               float* __restrict__ d, int n) {
  for (int i = blockIdx.x * blockDim.x + threadIdx.x; i < n; i += gridDim.x * blockDim.x)
    d[i] = a[i] + b[i];
}

__global__ void k_zero_f32(float* __restrict__ p, int n) {
  for (int i = blockIdx.x * blockDim.x + threadIdx.x; i < n; i += gridDim.x * blockDim.x)
    p[i] = 0.0f;
}
__global__ void k_zero_bf16(unsigned short* __restrict__ p, int n) {
  for (int i = blockIdx.x * blockDim.x + threadIdx.x; i < n; i += gridDim.x * blockDim.x)
    p[i] = 0;
}

// --------------------------- big projection GEMM ---------------------------
// C[M,N] (bf16) = A[M,K](bf16) @ W[N,K](bf16)^T + bias[N](f32)
// One wave computes a 32(M) x 64(N) tile: 8 WMMA per (2 A + 4 B) frag loads.
__global__ void __launch_bounds__(BDIM)
k_gemm32x64(const unsigned short* __restrict__ A, int K,
            const unsigned short* __restrict__ W,
            const float* __restrict__ bias,
            unsigned short* __restrict__ C, int N) {
  int lane   = threadIdx.x & 31;
  int wid    = blockIdx.x * (BDIM / 32) + (threadIdx.x >> 5);
  int tilesN = N >> 6;
  int tm = wid / tilesN;
  int tn = wid % tilesN;
  int row0 = tm << 5;
  int col0 = tn << 6;

  v8f acc[2][4] = {{v8f{0}, v8f{0}, v8f{0}, v8f{0}},
                   {v8f{0}, v8f{0}, v8f{0}, v8f{0}}};
  for (int k = 0; k < K; k += 32) {
    v16bf a0 = load_frag(A, K, row0,      k, lane);
    v16bf a1 = load_frag(A, K, row0 + 16, k, lane);
#pragma unroll
    for (int t = 0; t < 4; ++t) {
      v16bf b = load_frag(W, K, col0 + t * 16, k, lane);
      acc[0][t] = wmma_bf16(a0, b, acc[0][t]);
      acc[1][t] = wmma_bf16(a1, b, acc[1][t]);
    }
  }
  // C/D layout: lane holds (m = vgpr + 8*(lane>=16), n = lane&15)
  int n  = lane & 15;
  int mh = lane >> 4;
#pragma unroll
  for (int s = 0; s < 2; ++s) {
#pragma unroll
    for (int t = 0; t < 4; ++t) {
      int col = col0 + t * 16 + n;
      float bv = bias[col];
#pragma unroll
      for (int r = 0; r < 8; ++r) {
        int m = row0 + s * 16 + r + (mh << 3);
        C[(size_t)m * N + col] = f2bf(acc[s][t][r] + bv);
      }
    }
  }
}

// ------------------------------ LSTM timestep ------------------------------
// gates[64,4096] = xproj_t[64,4096] + h_{t-1}[64,1024] @ Whh[4096,1024]^T
// Wave pair (kh=0,1) splits K=1024 into two 512 halves for the same
// (16 batch x 16 hidden) tile; partials reduced through LDS. Each wave
// accumulates all four gate tiles i/f/g/o sharing one A fragment.
__global__ void __launch_bounds__(BDIM)
k_lstm_step(const unsigned short* __restrict__ xproj, // [64][4096] (bias folded)
            const unsigned short* __restrict__ hprev, // [64][1024]
            const unsigned short* __restrict__ Whh,   // [4096][1024]
            float* __restrict__ c,                    // [64][1024]
            unsigned short* __restrict__ hout) {      // [64][1024]
  const int H = 1024;
  __shared__ __align__(16) float red[4][4][32][8];    // [pair][gate][lane][r] = 16KB

  int lane = threadIdx.x & 31;
  int wave = threadIdx.x >> 5;    // 0..7
  int pair = wave >> 1;           // 0..3 : tile slot within block
  int kh   = wave & 1;            // K half
  int tile = blockIdx.x * 4 + pair;   // 0..255
  int tm = tile >> 6;             // batch tile (0..3)
  int tj = tile & 63;             // hidden tile (0..63)
  int row0 = tm << 4;
  int j0   = tj << 4;

  v8f acc[4] = {v8f{0}, v8f{0}, v8f{0}, v8f{0}};
  int k0 = kh << 9;               // 0 or 512
  for (int k = k0; k < k0 + 512; k += 32) {
    v16bf a = load_frag(hprev, H, row0, k, lane);
#pragma unroll
    for (int g = 0; g < 4; ++g) {
      v16bf b = load_frag(Whh, H, g * H + j0, k, lane);
      acc[g] = wmma_bf16(a, b, acc[g]);
    }
  }

  if (kh == 1) {
#pragma unroll
    for (int g = 0; g < 4; ++g)
      *(v8f*)&red[pair][g][lane][0] = acc[g];         // 2x ds_store_b128
  }
  __syncthreads();
  if (kh == 0) {
#pragma unroll
    for (int g = 0; g < 4; ++g) {
      v8f p = *(const v8f*)&red[pair][g][lane][0];    // 2x ds_load_b128
#pragma unroll
      for (int r = 0; r < 8; ++r) acc[g][r] += p[r];
    }
    int n  = lane & 15;
    int mh = lane >> 4;
#pragma unroll
    for (int r = 0; r < 8; ++r) {
      int m = row0 + r + (mh << 3);         // batch index
      int j = j0 + n;                       // hidden index
      const unsigned short* xp = xproj + (size_t)m * 4096;
      float gi = sigmoidf_(acc[0][r] + bf2f(xp[j]));
      float gf = sigmoidf_(acc[1][r] + bf2f(xp[1024 + j]));
      float gg = tanhf    (acc[2][r] + bf2f(xp[2048 + j]));
      float go = sigmoidf_(acc[3][r] + bf2f(xp[3072 + j]));
      float cn = gf * c[(size_t)m * H + j] + gi * gg;
      c[(size_t)m * H + j]    = cn;
      hout[(size_t)m * H + j] = f2bf(go * tanhf(cn));
    }
  }
}

// --------------------------------- FC head ---------------------------------
// out[b][t][o] = hseq1[t][b][:] . W_fc[o][:] + b_fc[o]   (fp32 output, permuted)
__global__ void __launch_bounds__(BDIM)
k_fc(const unsigned short* __restrict__ A,  // [T*B][1024], rows [t][b]
     const unsigned short* __restrict__ W,  // [256][1024]
     const float* __restrict__ bias,        // [256] fp32
     float* __restrict__ out) {             // [64][512][256]
  const int K = 1024, N = 256;
  int lane   = threadIdx.x & 31;
  int wid    = blockIdx.x * (BDIM / 32) + (threadIdx.x >> 5);
  int tilesN = N >> 6; // 4
  int tm = wid / tilesN;
  int tn = wid % tilesN;
  int row0 = tm << 5;
  int col0 = tn << 6;

  v8f acc[2][4] = {{v8f{0}, v8f{0}, v8f{0}, v8f{0}},
                   {v8f{0}, v8f{0}, v8f{0}, v8f{0}}};
  for (int k = 0; k < K; k += 32) {
    v16bf a0 = load_frag(A, K, row0,      k, lane);
    v16bf a1 = load_frag(A, K, row0 + 16, k, lane);
#pragma unroll
    for (int t = 0; t < 4; ++t) {
      v16bf b = load_frag(W, K, col0 + t * 16, k, lane);
      acc[0][t] = wmma_bf16(a0, b, acc[0][t]);
      acc[1][t] = wmma_bf16(a1, b, acc[1][t]);
    }
  }
  int n  = lane & 15;
  int mh = lane >> 4;
#pragma unroll
  for (int s = 0; s < 2; ++s) {
#pragma unroll
    for (int t = 0; t < 4; ++t) {
      int col = col0 + t * 16 + n;
      float bv = bias[col];
#pragma unroll
      for (int r = 0; r < 8; ++r) {
        int m  = row0 + s * 16 + r + (mh << 3);   // m = tstep*64 + b
        int b  = m & 63;
        int ts = m >> 6;
        out[((size_t)b * 512 + ts) * 256 + col] = acc[s][t][r] + bv;
      }
    }
  }
}

// ------------------------------- host driver -------------------------------

extern "C" void kernel_launch(void* const* d_in, const int* in_sizes, int n_in,
                              void* d_out, int out_size, void* d_ws, size_t ws_size,
                              hipStream_t stream) {
  (void)in_sizes; (void)n_in; (void)out_size; (void)ws_size;

  const float* x     = (const float*)d_in[0];
  const float* W_ih0 = (const float*)d_in[1];
  const float* W_hh0 = (const float*)d_in[2];
  const float* b_ih0 = (const float*)d_in[3];
  const float* b_hh0 = (const float*)d_in[4];
  const float* W_ih1 = (const float*)d_in[5];
  const float* W_hh1 = (const float*)d_in[6];
  const float* b_ih1 = (const float*)d_in[7];
  const float* b_hh1 = (const float*)d_in[8];
  const float* W_fc  = (const float*)d_in[9];
  const float* b_fc  = (const float*)d_in[10];
  float* out = (float*)d_out;

  char* ws = (char*)d_ws;
  size_t off = 0;
  auto alloc = [&](size_t bytes) -> void* {
    void* p = ws + off;
    off += (bytes + 255) & ~(size_t)255;
    return p;
  };

  unsigned short* wih0  = (unsigned short*)alloc((size_t)G4_ * IN_ * 2);
  unsigned short* whh0  = (unsigned short*)alloc((size_t)G4_ * H_  * 2);
  unsigned short* wih1  = (unsigned short*)alloc((size_t)G4_ * H_  * 2);
  unsigned short* whh1  = (unsigned short*)alloc((size_t)G4_ * H_  * 2);
  unsigned short* wfc   = (unsigned short*)alloc((size_t)OUT_ * H_ * 2);
  float*          bias0 = (float*)alloc((size_t)G4_ * 4);
  float*          bias1 = (float*)alloc((size_t)G4_ * 4);
  unsigned short* xb    = (unsigned short*)alloc((size_t)M_ * IN_ * 2); // time-major
  unsigned short* xproj = (unsigned short*)alloc((size_t)M_ * G4_ * 2); // reused L0/L1
  unsigned short* hseq0 = (unsigned short*)alloc((size_t)M_ * H_ * 2);
  unsigned short* hseq1 = (unsigned short*)alloc((size_t)M_ * H_ * 2);
  float*          cbuf  = (float*)alloc((size_t)B_ * H_ * 4);
  unsigned short* h0    = (unsigned short*)alloc((size_t)B_ * H_ * 2);

  const int CVT_BLOCKS = 2048;

  // 1) weight/bias/input conversion to bf16 (round-to-nearest-even)
  k_cvt<<<CVT_BLOCKS, BDIM, 0, stream>>>(W_ih0, wih0, G4_ * IN_);
  k_cvt<<<CVT_BLOCKS, BDIM, 0, stream>>>(W_hh0, whh0, G4_ * H_);
  k_cvt<<<CVT_BLOCKS, BDIM, 0, stream>>>(W_ih1, wih1, G4_ * H_);
  k_cvt<<<CVT_BLOCKS, BDIM, 0, stream>>>(W_hh1, whh1, G4_ * H_);
  k_cvt<<<CVT_BLOCKS, BDIM, 0, stream>>>(W_fc,  wfc,  OUT_ * H_);
  k_cvt_x_tmajor<<<CVT_BLOCKS, BDIM, 0, stream>>>(x, xb);
  k_bias_combine<<<16, BDIM, 0, stream>>>(b_ih0, b_hh0, bias0, G4_);
  k_bias_combine<<<16, BDIM, 0, stream>>>(b_ih1, b_hh1, bias1, G4_);
  k_zero_bf16<<<64, BDIM, 0, stream>>>(h0, B_ * H_);
  k_zero_f32 <<<64, BDIM, 0, stream>>>(cbuf, B_ * H_);

  // 2) layer-0 input projection: xproj = xb @ W_ih0^T + bias0   [M, 4096] bf16
  {
    int waves  = (M_ / 32) * (G4_ / 64);
    int blocks = waves / (BDIM / 32);
    k_gemm32x64<<<blocks, BDIM, 0, stream>>>(xb, IN_, wih0, bias0, xproj, G4_);
  }

  // 3) layer-0 recurrent scan (512 sequential step kernels, split-K=2)
  for (int t = 0; t < T_; ++t) {
    const unsigned short* hp = (t == 0) ? h0 : (hseq0 + (size_t)(t - 1) * B_ * H_);
    k_lstm_step<<<64, BDIM, 0, stream>>>(xproj + (size_t)t * B_ * G4_, hp, whh0,
                                         cbuf, hseq0 + (size_t)t * B_ * H_);
  }

  // 4) layer-1 input projection: xproj = hseq0 @ W_ih1^T + bias1
  {
    int waves  = (M_ / 32) * (G4_ / 64);
    int blocks = waves / (BDIM / 32);
    k_gemm32x64<<<blocks, BDIM, 0, stream>>>(hseq0, H_, wih1, bias1, xproj, G4_);
  }
  k_zero_f32<<<64, BDIM, 0, stream>>>(cbuf, B_ * H_);

  // 5) layer-1 recurrent scan
  for (int t = 0; t < T_; ++t) {
    const unsigned short* hp = (t == 0) ? h0 : (hseq1 + (size_t)(t - 1) * B_ * H_);
    k_lstm_step<<<64, BDIM, 0, stream>>>(xproj + (size_t)t * B_ * G4_, hp, whh1,
                                         cbuf, hseq1 + (size_t)t * B_ * H_);
  }

  // 6) FC head with [T][B] -> [B][T] permuted fp32 store
  {
    int waves  = (M_ / 32) * (OUT_ / 64);
    int blocks = waves / (BDIM / 32);
    k_fc<<<blocks, BDIM, 0, stream>>>(hseq1, wfc, b_fc, out);
  }
}
